// RNNEncoder_24215025615109
// MI455X (gfx1250) — matrix operand order repaired and checked by reading, not compiled, not run on hardware
//
#include <hip/hip_runtime.h>
#include <hip/hip_bf16.h>

// ---------------------------------------------------------------------------
// ConvLSTM encoder for MI455X (gfx1250), wave32 + WMMA bf16.
// B=64, T=4, 16 pixels, HID=512, G=2048 gate channels.
// conv3x3 == 9 tap-GEMMs: M=2048, N=16 (pixels), K=Cin chunked by 32.
// 2 batches per WG (16 waves): each weight A-fragment feeds 2 WMMAs.
// Zero-halo row in LDS removes per-lane tap predication.
// A-fragments stream through a depth-3 register ring (pipelined loads).
// x_t tiles staged to LDS with global_load_async_to_lds_b128 (ASYNCcnt).
// ---------------------------------------------------------------------------

typedef __bf16 v16bf __attribute__((ext_vector_type(16)));
typedef float  v8f   __attribute__((ext_vector_type(8)));

union FragU {
    v16bf v;
    uint4 q[2];
    unsigned short h[16];
};

__device__ __forceinline__ unsigned short f2bf(float f) {
    unsigned int u = __float_as_uint(f);
    unsigned int r = u + 0x7FFFu + ((u >> 16) & 1u);   // round-to-nearest-even
    return (unsigned short)(r >> 16);
}
__device__ __forceinline__ float bf2f(unsigned short h) {
    return __uint_as_float(((unsigned int)h) << 16);
}
__device__ __forceinline__ unsigned int pack2bf(float a, float b) {
    return (unsigned int)f2bf(a) | ((unsigned int)f2bf(b) << 16);
}

#define TBP 256       // prep kernels
#define TBL 512       // layer kernel: 16 wave32
#define NPIX 16
#define HID 512
#define GCH 2048
#define MT_PER_WAVE 8 // 16 waves x 8 mtiles = 128 tiles = 2048 channels
#define NB 2          // batches per workgroup

// --------------------------- small prep kernels ----------------------------

__global__ void action_mlp_kernel(const float* __restrict__ act,
                                  const float* __restrict__ w1, const float* __restrict__ b1,
                                  const float* __restrict__ w2, const float* __restrict__ b2,
                                  float* __restrict__ aemb) {
    const int b = threadIdx.x;
    if (b >= 64) return;
    const float a0 = act[b*3+0], a1 = act[b*3+1], a2 = act[b*3+2];
    float h[8];
    #pragma unroll
    for (int j = 0; j < 8; ++j) {
        float s = a0*w1[0*8+j] + a1*w1[1*8+j] + a2*w1[2*8+j] + b1[j];
        h[j] = s > 0.f ? s : 0.f;
    }
    #pragma unroll
    for (int k = 0; k < 16; ++k) {
        float s = b2[k];
        #pragma unroll
        for (int j = 0; j < 8; ++j) s += h[j]*w2[j*16+k];
        aemb[b*16+k] = s;
    }
}

// X0[b][t][p][c], c in [0,544): c<512 embedding, c==512 action channel, pad 0.
__global__ void build_x0_kernel(const float* __restrict__ emb,
                                const float* __restrict__ aemb,
                                unsigned short* __restrict__ X0) {
    const int bt = blockIdx.x;      // b*4 + t
    const int b  = bt >> 2;
    for (int i = threadIdx.x; i < NPIX*544; i += TBP) {
        const int p = i / 544;
        const int c = i % 544;
        float v;
        if (c < 512)       v = emb[((size_t)bt*512 + c)*NPIX + p];
        else if (c == 512) v = aemb[b*16 + p];
        else               v = 0.f;
        X0[(size_t)bt*NPIX*544 + i] = f2bf(v);
    }
}

// Pack fp32 conv weights [G][Cin][3][3] into bf16 WMMA A-fragments:
// frags[tap][kc][mtile][lane*16 + e]  (contiguous k order: k = 16*(lane/16) + e, same as B)
__global__ void pack_weights_kernel(const float* __restrict__ w,
                                    unsigned short* __restrict__ frags,
                                    int Cin, int Kc) {
    const int f     = blockIdx.x;           // fragment id
    const int tap   = f / (Kc*128);
    const int rem   = f % (Kc*128);
    const int kc    = rem / 128;
    const int mtile = rem % 128;
    for (int hIdx = threadIdx.x; hIdx < 512; hIdx += TBP) {
        const int lane = hIdx >> 4;
        const int e    = hIdx & 15;
        const int m    = lane & 15;
        const int klo  = (lane < 16) ? 0 : 16;
        const int K    = klo + e;
        const int gch  = mtile*16 + m;
        const int c    = kc*32 + K;
        float v = (c < Cin) ? w[((size_t)gch*Cin + c)*9 + tap] : 0.f;
        frags[(size_t)f*512 + hIdx] = f2bf(v);
    }
}

// ------------------------------ core kernel --------------------------------

// 9-tap conv as WMMA GEMM. Two B operands (two batches) share every A
// fragment. A fragments prefetched with a depth-3 ring (flat f = kc*8+mt).
__device__ __forceinline__ void conv_gemm(v8f acc[MT_PER_WAVE*NB],
                                          const unsigned short* __restrict__ lds0,
                                          const unsigned short* __restrict__ lds1,
                                          int rowStride, int Kc,
                                          const unsigned short* __restrict__ frag,
                                          int wave, int lane, int py, int px, int kbase) {
    for (int tap = 0; tap < 9; ++tap) {
        const int yy = py + tap/3 - 1;
        const int xx = px + tap%3 - 1;
        const bool valid = ((unsigned)yy < 4u) && ((unsigned)xx < 4u);
        const int spRow = valid ? (yy*4 + xx) : 16;      // row 16 = zero halo
        const unsigned short* bbase0 = lds0 + (size_t)spRow*rowStride + kbase;
        const unsigned short* bbase1 = lds1 + (size_t)spRow*rowStride + kbase;
        const unsigned short* wf0 =
            frag + ((size_t)(tap*Kc)*128 + (size_t)wave*MT_PER_WAVE)*512 + (size_t)lane*16;

        FragU ar[4];
        #pragma unroll
        for (int i0 = 0; i0 < 3; ++i0) {                 // preload f = 0,1,2
            const uint4* a4 = (const uint4*)(wf0 + (size_t)i0*512);
            ar[i0].q[0] = a4[0]; ar[i0].q[1] = a4[1];
        }

        for (int kc = 0; kc < Kc; ++kc) {
            FragU bf0, bf1;
            {
                const uint4* q0 = (const uint4*)(bbase0 + (size_t)kc*32);
                bf0.q[0] = q0[0]; bf0.q[1] = q0[1];      // unconditional ds_load_b128
                const uint4* q1 = (const uint4*)(bbase1 + (size_t)kc*32);
                bf1.q[0] = q1[0]; bf1.q[1] = q1[1];
            }
            #pragma unroll
            for (int mt = 0; mt < MT_PER_WAVE; ++mt) {
                // prefetch fragment f+3 (crosses kc boundary for mt >= 5)
                const int kcn = kc + ((mt + 3) >> 3);
                if (kcn < Kc) {
                    const uint4* a4 = (const uint4*)
                        (wf0 + (size_t)kcn*(128*512) + (size_t)((mt + 3) & 7)*512);
                    ar[(mt + 3) & 3].q[0] = a4[0];
                    ar[(mt + 3) & 3].q[1] = a4[1];
                }
                acc[mt*NB + 0] = __builtin_amdgcn_wmma_f32_16x16x32_bf16(
                    false, ar[mt & 3].v, false, bf0.v, (short)0, acc[mt*NB + 0], false, false);
                acc[mt*NB + 1] = __builtin_amdgcn_wmma_f32_16x16x32_bf16(
                    false, ar[mt & 3].v, false, bf1.v, (short)0, acc[mt*NB + 1], false, false);
            }
        }
    }
}

__global__ __launch_bounds__(TBL)
void convlstm_layer_kernel(const unsigned short* __restrict__ Xin,   // [B][T][16][CinPad]
                           const unsigned short* __restrict__ wxf_f,
                           const unsigned short* __restrict__ wxf_b,
                           const unsigned short* __restrict__ whf_f,
                           const unsigned short* __restrict__ whf_b,
                           const float* __restrict__ bias_f,
                           const float* __restrict__ bias_b,
                           unsigned short* __restrict__ Hout,        // [B][T][16][1024] or null
                           float* __restrict__ OutMean,              // [B][1024][16] or null
                           int CinPad, int Kc) {
    extern __shared__ char smem[];
    const int b0   = blockIdx.x * NB;
    const int dir  = blockIdx.y;     // 0 = forward, 1 = backward
    const int tid  = threadIdx.x;
    const int lane = tid & 31;
    const int wave = tid >> 5;

    // LDS: [2][17][CinPad] x (halo row 16 = zeros), [2][17][512] h,
    //      [2][16][512] f32 cell, [2][16][2048] bf16 gates
    const int xTile = 17*CinPad;               // halves per batch x tile
    const int hTile = 17*HID;
    unsigned short* xlds   = (unsigned short*)smem;
    unsigned short* hlds   = xlds + NB*xTile;
    float*          clds   = (float*)(hlds + NB*hTile);
    unsigned short* glds16 = (unsigned short*)(clds + NB*NPIX*HID);

    const unsigned short* wxfrag = dir ? wxf_b : wxf_f;
    const unsigned short* whfrag = dir ? whf_b : whf_f;
    const float*          bias   = dir ? bias_b : bias_f;

    for (int i = tid; i < NB*hTile; i += TBL) hlds[i] = 0;          // incl. halo
    for (int i = tid; i < NB*NPIX*HID; i += TBL) clds[i] = 0.f;
    for (int i = tid; i < NB*CinPad; i += TBL) {                    // x halo rows
        const int bb = i / CinPad, c = i % CinPad;
        xlds[bb*xTile + 16*CinPad + c] = 0;
    }
    __syncthreads();

    // per-lane WMMA B geometry: column N == pixel
    const int nn = lane & 15;
    const int py = nn >> 2, px = nn & 3;
    const int kbase = (lane < 16) ? 0 : 16;

    float outacc[32];
    #pragma unroll
    for (int j = 0; j < 32; ++j) outacc[j] = 0.f;

    const int n16 = (NPIX*CinPad) >> 3;        // 16-byte chunks per x tile
    const int bsel = tid >> 8;                 // pointwise batch half
    const int tl   = tid & 255;

    for (int ts = 0; ts < 4; ++ts) {
        const int t = dir ? (3 - ts) : ts;

        // Stage both x_t tiles into LDS via async DMA (ASYNCcnt-tracked).
        #pragma unroll
        for (int bb = 0; bb < NB; ++bb) {
            const unsigned short* src = Xin + ((size_t)((b0 + bb)*4 + t)*NPIX)*CinPad;
            const unsigned int dbase = (unsigned int)(bb*xTile)*2u;
            for (int i = tid; i < n16; i += TBL) {
                unsigned int ldsOff = dbase + (unsigned int)i*16u;
                unsigned long long ga = (unsigned long long)(src) + (unsigned long long)i*16ull;
                asm volatile("global_load_async_to_lds_b128 %0, %1, off"
                             :: "v"(ldsOff), "v"(ga) : "memory");
            }
        }
        asm volatile("s_wait_asynccnt 0x0" ::: "memory");
        __syncthreads();

        v8f acc[MT_PER_WAVE*NB];
        #pragma unroll
        for (int q = 0; q < MT_PER_WAVE*NB; ++q)
            acc[q] = v8f{0.f,0.f,0.f,0.f,0.f,0.f,0.f,0.f};

        conv_gemm(acc, xlds, xlds + xTile, CinPad, Kc, wxfrag,
                  wave, lane, py, px, kbase);                       // wx * x_t
        conv_gemm(acc, hlds, hlds + hTile, HID, 16, whfrag,
                  wave, lane, py, px, kbase);                       // wh * h_{t-1}

        {   // scatter C-fragments to bf16 gates LDS, packed ds_store_b128
            const int p     = lane & 15;
            const int mrow0 = (lane < 16) ? 0 : 8;
            #pragma unroll
            for (int mt = 0; mt < MT_PER_WAVE; ++mt) {
                #pragma unroll
                for (int bb = 0; bb < NB; ++bb) {
                    const v8f a = acc[mt*NB + bb];
                    uint4 wv;
                    wv.x = pack2bf(a[0], a[1]);
                    wv.y = pack2bf(a[2], a[3]);
                    wv.z = pack2bf(a[4], a[5]);
                    wv.w = pack2bf(a[6], a[7]);
                    *(uint4*)(glds16 + (size_t)bb*NPIX*GCH
                              + (size_t)p*GCH + wave*128 + mt*16 + mrow0) = wv;
                }
            }
        }
        __syncthreads();

        // LSTM pointwise: threads [0,256) -> batch0, [256,512) -> batch1.
        {
            const unsigned short* gl = glds16 + (size_t)bsel*NPIX*GCH;
            float* cl = clds + (size_t)bsel*NPIX*HID;
            unsigned short* hl = hlds + (size_t)bsel*hTile;
            #pragma unroll
            for (int j = 0; j < 32; ++j) {
                const int linear = j*256 + tl;
                const int p  = linear >> 9;        // 0..15
                const int hc = linear & 511;       // 0..511
                const float ig = bf2f(gl[(size_t)p*GCH + hc        ]) + bias[hc];
                const float fg = bf2f(gl[(size_t)p*GCH + hc +  HID ]) + bias[hc +  HID];
                const float gg = bf2f(gl[(size_t)p*GCH + hc + 2*HID]) + bias[hc + 2*HID];
                const float og = bf2f(gl[(size_t)p*GCH + hc + 3*HID]) + bias[hc + 3*HID];
                const float si = 1.f/(1.f + __expf(-ig));
                const float sf = 1.f/(1.f + __expf(-fg));
                const float so = 1.f/(1.f + __expf(-og));
                const float c  = sf*cl[p*HID + hc] + si*tanhf(gg);
                cl[p*HID + hc] = c;
                const float hv = so*tanhf(c);
                hl[p*HID + hc] = f2bf(hv);         // pixel-major for next-t B frags
                outacc[j] += hv;
                if (Hout)
                    Hout[(((size_t)((b0 + bsel)*4 + t)*NPIX) + p)*1024 + dir*HID + hc]
                        = f2bf(hv);
            }
        }
        __syncthreads();
    }

    if (OutMean) {
        #pragma unroll
        for (int j = 0; j < 32; ++j) {
            const int linear = j*256 + tl;
            const int p  = linear >> 9;
            const int hc = linear & 511;
            OutMean[((size_t)(b0 + bsel)*1024 + dir*HID + hc)*NPIX + p] = outacc[j]*0.25f;
        }
    }
}

// ------------------------------- launcher ----------------------------------

extern "C" void kernel_launch(void* const* d_in, const int* in_sizes, int n_in,
                              void* d_out, int out_size, void* d_ws, size_t ws_size,
                              hipStream_t stream) {
    (void)in_sizes; (void)n_in; (void)out_size; (void)ws_size;

    const float* emb  = (const float*)d_in[0];
    const float* act  = (const float*)d_in[1];
    /* d_in[2] embedding_length: all == T, unused */
    const float* w1   = (const float*)d_in[3];
    const float* b1   = (const float*)d_in[4];
    const float* w2   = (const float*)d_in[5];
    const float* b2   = (const float*)d_in[6];
    const float* wx0f = (const float*)d_in[7];
    const float* wh0f = (const float*)d_in[8];
    const float* bg0f = (const float*)d_in[9];
    const float* wx0b = (const float*)d_in[10];
    const float* wh0b = (const float*)d_in[11];
    const float* bg0b = (const float*)d_in[12];
    const float* wx1f = (const float*)d_in[13];
    const float* wh1f = (const float*)d_in[14];
    const float* bg1f = (const float*)d_in[15];
    const float* wx1b = (const float*)d_in[16];
    const float* wh1b = (const float*)d_in[17];
    const float* bg1b = (const float*)d_in[18];

    char* ws = (char*)d_ws;
    size_t off = 0;
    auto carve = [&](size_t bytes) -> char* {
        char* p = ws + off;
        off = (off + bytes + 255) & ~(size_t)255;
        return p;
    };
    float*          aemb  = (float*)         carve((size_t)64*16*4);
    unsigned short* X0    = (unsigned short*)carve((size_t)64*4*16*544*2);
    unsigned short* X1    = (unsigned short*)carve((size_t)64*4*16*1024*2);
    unsigned short* fwx0f = (unsigned short*)carve((size_t)9*17*128*512*2);
    unsigned short* fwx0b = (unsigned short*)carve((size_t)9*17*128*512*2);
    unsigned short* fwh0f = (unsigned short*)carve((size_t)9*16*128*512*2);
    unsigned short* fwh0b = (unsigned short*)carve((size_t)9*16*128*512*2);
    unsigned short* fwx1f = (unsigned short*)carve((size_t)9*32*128*512*2);
    unsigned short* fwx1b = (unsigned short*)carve((size_t)9*32*128*512*2);
    unsigned short* fwh1f = (unsigned short*)carve((size_t)9*16*128*512*2);
    unsigned short* fwh1b = (unsigned short*)carve((size_t)9*16*128*512*2);

    // prep
    action_mlp_kernel<<<1, 64, 0, stream>>>(act, w1, b1, w2, b2, aemb);
    build_x0_kernel<<<64*4, TBP, 0, stream>>>(emb, aemb, X0);

    pack_weights_kernel<<<9*17*128, TBP, 0, stream>>>(wx0f, fwx0f,  513, 17);
    pack_weights_kernel<<<9*17*128, TBP, 0, stream>>>(wx0b, fwx0b,  513, 17);
    pack_weights_kernel<<<9*16*128, TBP, 0, stream>>>(wh0f, fwh0f,  512, 16);
    pack_weights_kernel<<<9*16*128, TBP, 0, stream>>>(wh0b, fwh0b,  512, 16);
    pack_weights_kernel<<<9*32*128, TBP, 0, stream>>>(wx1f, fwx1f, 1024, 32);
    pack_weights_kernel<<<9*32*128, TBP, 0, stream>>>(wx1b, fwx1b, 1024, 32);
    pack_weights_kernel<<<9*16*128, TBP, 0, stream>>>(wh1f, fwh1f,  512, 16);
    pack_weights_kernel<<<9*16*128, TBP, 0, stream>>>(wh1b, fwh1b,  512, 16);

    // LDS bytes: 2*(17*CinPad + 17*512)*2 + 2*16*512*4 + 2*16*2048*2
    const int smem0 = 2*(17*544  + 17*512)*2 + 2*16*512*4 + 2*16*2048*2;  // 268416
    const int smem1 = 2*(17*1024 + 17*512)*2 + 2*16*512*4 + 2*16*2048*2;  // 301056
    (void)hipFuncSetAttribute((const void*)convlstm_layer_kernel,
                              hipFuncAttributeMaxDynamicSharedMemorySize, smem1);

    // layer 0 (both directions in one launch; disjoint halves of X1)
    convlstm_layer_kernel<<<dim3(32, 2), TBL, smem0, stream>>>(
        X0, fwx0f, fwx0b, fwh0f, fwh0b, bg0f, bg0b,
        X1, (float*)nullptr, 544, 17);

    // layer 1 -> time-mean directly into d_out
    convlstm_layer_kernel<<<dim3(32, 2), TBL, smem1, stream>>>(
        X1, fwx1f, fwx1b, fwh1f, fwh1b, bg1f, bg1b,
        (unsigned short*)nullptr, (float*)d_out, 1024, 32);
}
